// IntraMA_73263552135519
// MI455X (gfx1250) — matrix operand order repaired and checked by reading, not compiled, not run on hardware
//
#include <hip/hip_runtime.h>

typedef __attribute__((ext_vector_type(16))) _Float16 v16h;
typedef __attribute__((ext_vector_type(8)))  _Float16 v8h;
typedef __attribute__((ext_vector_type(4)))  _Float16 v4h;
typedef __attribute__((ext_vector_type(8)))  float    v8f;
typedef __attribute__((ext_vector_type(4)))  float    v4f;

#define BB 256
#define NN 512
#define PP 7
#define DD 128
#define HH 4
#define HD 32
#define BN (BB*NN)
#define GROUP 16   // sequences per block (2 seqs per 16-row token tile)

// Padded LDS row strides (in elements) so consecutive rows shift by 4 banks
// instead of aliasing to the same 64-bank pattern.
#define XS 136   // X row stride (halves)       : 272 B
#define QS 392   // QKV row stride (halves)     : 784 B
#define OS 136   // OutRows row stride (halves) : 272 B
#define PS 132   // Proj row stride (floats)    : 528 B

// ---------------------------------------------------------------------------
// A-matrix fragment (16x32 f16) streamed from a global f32 matrix W[M][K]
// (row-major, ld = K).  Per ISA A layout: lane L holds row M = L&15;
//   elements 0..7  = K (kbase + koffA)..+7        koffA = (L>=16) ? 8 : 0
//   elements 8..15 = K (kbase + 16 + koffA)..+7
__device__ __forceinline__ v16h load_fragA_gf32(const float* __restrict__ W, int ld,
                                                int mrow, int kbase, int koffA) {
  const float* p = W + (size_t)mrow * ld + kbase + koffA;
  v4f f0 = *(const v4f*)(p + 0);
  v4f f1 = *(const v4f*)(p + 4);
  v4f f2 = *(const v4f*)(p + 16);
  v4f f3 = *(const v4f*)(p + 20);
  v16h r;
#pragma unroll
  for (int i = 0; i < 4; ++i) {
    r[i]      = (_Float16)f0[i];
    r[4 + i]  = (_Float16)f1[i];
    r[8 + i]  = (_Float16)f2[i];
    r[12 + i] = (_Float16)f3[i];
  }
  return r;
}

// ---------------------------------------------------------------------------
// Kernel 1: per-sequence MHA (WMMA GEMMs + tiny VALU attention) + gating logits.
// Block = 256 threads (8 waves), GROUP=16 sequences.
// GEMMs are computed transposed (C[feature][row]) so each lane's D registers
// hold 8 consecutive features of one row -> single b128 LDS store per tile.
__global__ __launch_bounds__(256)
void intrama_k1(const float* __restrict__ inst,
                const float* __restrict__ pos_emb,
                const float* __restrict__ in_w,  const float* __restrict__ in_b,
                const float* __restrict__ out_w, const float* __restrict__ out_b,
                const float* __restrict__ attn1_w, const float* __restrict__ attn2,
                float* __restrict__ wsA, float* __restrict__ wsMP) {
  __shared__ __align__(16) _Float16 s_QKV[128 * QS];            // ~98 KB
  __shared__ __align__(16) _Float16 s_OutRows[2 * GROUP * OS];  // ~8.5 KB
  __shared__ float s_scores[GROUP][8][8];                       // 4 KB
  __shared__ float s_attn1[HH * DD];
  __shared__ float s_attn2[DD];
  __shared__ float s_bin[3 * DD];
  __shared__ float s_bout[DD];
  // X (f16, 128 x XS = 34 KB) overlaid with Proj (f32, 32 x PS = 16.5 KB)
  __shared__ __align__(16) char s_xp_raw[128 * XS * sizeof(_Float16)];
  _Float16* s_X    = (_Float16*)s_xp_raw;
  float*    s_Proj = (float*)s_xp_raw;

  const int tid   = threadIdx.x;
  const int lane  = tid & 31;
  const int wave  = tid >> 5;
  const int nl    = lane & 15;
  const int hs    = lane >> 4;   // half-wave select
  const int koffA = hs * 8;
  const int koffB = hs * 8;
  const int g0    = blockIdx.x * GROUP;  // first flat sequence index (bn)

  // ---- Phase A: constants to LDS ----
  for (int i = tid; i < HH * DD; i += 256) s_attn1[i] = attn1_w[i];
  for (int i = tid; i < DD; i += 256) { s_attn2[i] = attn2[i]; s_bout[i] = out_b[i]; }
  for (int i = tid; i < 3 * DD; i += 256) s_bin[i] = in_b[i];

  // ---- Phase B: X = instances + pos_emb (f32 -> f16), float4 granularity ----
  // seq s -> token tile s/2, row (s&1)*8 + p ; rows p==7 are zero padding.
  for (int idx = tid; idx < GROUP * PP * (DD / 4); idx += 256) {
    int s  = idx / (PP * (DD / 4));
    int r  = idx - s * (PP * (DD / 4));
    int p  = r / (DD / 4);
    int d4 = r - p * (DD / 4);
    v4f v  = *(const v4f*)(inst + ((size_t)(g0 + s) * PP + p) * DD + d4 * 4);
    v4f pe = *(const v4f*)(pos_emb + p * DD + d4 * 4);
    int row = (s >> 1) * 16 + (s & 1) * 8 + p;
    v4h hv;
#pragma unroll
    for (int i = 0; i < 4; ++i) hv[i] = (_Float16)(v[i] + pe[i]);
    *(v4h*)(s_X + row * XS + d4 * 4) = hv;
  }
  for (int idx = tid; idx < 16 * (DD / 4); idx += 256) {  // zero pad rows
    int t  = idx / (DD / 4);
    int d4 = idx - t * (DD / 4);
    v4h z = {};
    *(v4h*)(s_X + (t * 8 + 7) * XS + d4 * 4) = z;
  }
  __syncthreads();

  // ---- Phase C: QKV[row][f] = Win[f][:] . X[row][:] + bin[f] ----
  // A = Win (wave owns feature tiles 3w..3w+2, fragments held in registers),
  // B = X token rows from LDS (contiguous per lane), M=384, N=128, K=128.
  {
    v16h af[3][4];
    v8f  cbias[3];                 // bias-initialized accumulators, hoisted
#pragma unroll
    for (int i = 0; i < 3; ++i) {
      const int f0 = (wave * 3 + i) * 16 + hs * 8;
#pragma unroll
      for (int r = 0; r < 8; ++r) cbias[i][r] = s_bin[f0 + r];
#pragma unroll
      for (int kk = 0; kk < 4; ++kk)
        af[i][kk] = load_fragA_gf32(in_w, DD, (wave * 3 + i) * 16 + nl, kk * 32, koffA);
    }
#pragma unroll
    for (int ntile = 0; ntile < 8; ++ntile) {
      const int row = ntile * 16 + nl;   // this lane's token row (B: N = L&15)
      v16h bf[4];
#pragma unroll
      for (int kk = 0; kk < 4; ++kk) {
        v8h b0 = *(const v8h*)(s_X + row * XS + kk * 32 + koffB);
        v8h b1 = *(const v8h*)(s_X + row * XS + kk * 32 + 16 + koffB);
        bf[kk] = __builtin_shufflevector(b0, b1, 0,1,2,3,4,5,6,7,8,9,10,11,12,13,14,15);
      }
#pragma unroll
      for (int i = 0; i < 3; ++i) {
        const int f0 = (wave * 3 + i) * 16 + hs * 8;  // D: M = r + hs*8 within tile
        v8f c = cbias[i];
#pragma unroll
        for (int kk = 0; kk < 4; ++kk)
          c = __builtin_amdgcn_wmma_f32_16x16x32_f16(false, af[i][kk], false, bf[kk],
                                                     (short)0, c, false, false);
        v8h hv;
#pragma unroll
        for (int r = 0; r < 8; ++r) hv[r] = (_Float16)c[r];
        *(v8h*)(s_QKV + row * QS + f0) = hv;   // one b128 store per tile
      }
    }
  }
  __syncthreads();

  // ---- Phase D: attention (only query rows p=0 and p=6 matter) ----
  {
    const int sseq  = wave * 2 + hs;        // this half-wave's sequence
    const int rbase = wave * 16 + hs * 8;   // QKV row of token p=0
    for (int t = nl; t < HH * 2 * PP; t += 16) {
      int h = t / (2 * PP);
      int rem = t - h * 2 * PP;
      int i = rem / PP;
      int j = rem - i * PP;
      const _Float16* q = s_QKV + (rbase + (i ? 6 : 0)) * QS + h * HD;
      const _Float16* k = s_QKV + (rbase + j) * QS + DD + h * HD;
      float acc = 0.f;
#pragma unroll
      for (int d = 0; d < HD; ++d) acc += (float)q[d] * (float)k[d];
      s_scores[sseq][h * 2 + i][j] = acc * 0.17677669529663687f;  // 1/sqrt(32)
    }
    __syncthreads();
    for (int hi = 0; hi < HH * 2; ++hi) {
      int h = hi >> 1;
      float sc[PP];
      float mx = -3.0e38f;
#pragma unroll
      for (int j = 0; j < PP; ++j) { sc[j] = s_scores[sseq][hi][j]; mx = fmaxf(mx, sc[j]); }
      float se = 0.f;
#pragma unroll
      for (int j = 0; j < PP; ++j) { sc[j] = __expf(sc[j] - mx); se += sc[j]; }
      float inv = 1.f / se;
#pragma unroll
      for (int dd = 0; dd < 2; ++dd) {
        int d = nl + dd * 16;
        float acc = 0.f;
#pragma unroll
        for (int j = 0; j < PP; ++j)
          acc += sc[j] * (float)s_QKV[(rbase + j) * QS + 2 * DD + h * HD + d];
        s_OutRows[(sseq * 2 + (hi & 1)) * OS + h * HD + d] = (_Float16)(acc * inv);
      }
    }
  }
  __syncthreads();

  // ---- Phase E: Proj[row][f] = Wout[f][:] . OutRows[row][:] + bout[f] ----
  // A = Wout (wave owns feature tile w), B = OutRows (2 N-tiles). M=128,N=32,K=128.
  {
    v16h af[4];
#pragma unroll
    for (int kk = 0; kk < 4; ++kk)
      af[kk] = load_fragA_gf32(out_w, DD, wave * 16 + nl, kk * 32, koffA);
    const int f0 = wave * 16 + hs * 8;
    v8f cbias;
#pragma unroll
    for (int r = 0; r < 8; ++r) cbias[r] = s_bout[f0 + r];
#pragma unroll
    for (int ntile = 0; ntile < 2; ++ntile) {
      const int row = ntile * 16 + nl;
      v16h bf[4];
#pragma unroll
      for (int kk = 0; kk < 4; ++kk) {
        v8h b0 = *(const v8h*)(s_OutRows + row * OS + kk * 32 + koffB);
        v8h b1 = *(const v8h*)(s_OutRows + row * OS + kk * 32 + 16 + koffB);
        bf[kk] = __builtin_shufflevector(b0, b1, 0,1,2,3,4,5,6,7,8,9,10,11,12,13,14,15);
      }
      v8f c = cbias;
#pragma unroll
      for (int kk = 0; kk < 4; ++kk)
        c = __builtin_amdgcn_wmma_f32_16x16x32_f16(false, af[kk], false, bf[kk],
                                                   (short)0, c, false, false);
      v4f lo = {c[0], c[1], c[2], c[3]};
      v4f hi4 = {c[4], c[5], c[6], c[7]};
      *(v4f*)(s_Proj + row * PS + f0)     = lo;   // Proj overlays X (safe after barrier)
      *(v4f*)(s_Proj + row * PS + f0 + 4) = hi4;
    }
  }
  __syncthreads();

  // ---- Phase F: a = leaky_relu(start@attn1^T + sum(mp*attn2)); store a & mp ----
  {
    const int s = tid >> 4;
    const int l = tid & 15;
    const float* startRow = s_Proj + (s * 2 + 0) * PS;
    const float* mpRow    = s_Proj + (s * 2 + 1) * PS;
    float pa0 = 0.f, pa1 = 0.f, pa2 = 0.f, pa3 = 0.f;
#pragma unroll
    for (int d = l; d < DD; d += 16) {
      float sv = startRow[d];
      float mv = mpRow[d];
      pa0 += sv * s_attn1[0 * DD + d];
      pa1 += sv * s_attn1[1 * DD + d];
      pa2 += sv * s_attn1[2 * DD + d];
      pa3 += sv * s_attn1[3 * DD + d];
      float a2c = mv * s_attn2[d];
      int h = d >> 5;
      if      (h == 0) pa0 += a2c;
      else if (h == 1) pa1 += a2c;
      else if (h == 2) pa2 += a2c;
      else             pa3 += a2c;
    }
#pragma unroll
    for (int off = 8; off; off >>= 1) {
      pa0 += __shfl_xor(pa0, off, 16);
      pa1 += __shfl_xor(pa1, off, 16);
      pa2 += __shfl_xor(pa2, off, 16);
      pa3 += __shfl_xor(pa3, off, 16);
    }
    const size_t bn = (size_t)(g0 + s);
    if (l == 0) {   // wsA stored transposed: [h][bn] for coalesced K2 reads
      wsA[0 * (size_t)BN + bn] = pa0 > 0.f ? pa0 : 0.01f * pa0;
      wsA[1 * (size_t)BN + bn] = pa1 > 0.f ? pa1 : 0.01f * pa1;
      wsA[2 * (size_t)BN + bn] = pa2 > 0.f ? pa2 : 0.01f * pa2;
      wsA[3 * (size_t)BN + bn] = pa3 > 0.f ? pa3 : 0.01f * pa3;
    }
  }
  // mp writeback, b128 granularity, coalesced
  for (int idx = tid; idx < GROUP * (DD / 4); idx += 256) {
    int s  = idx >> 5;
    int d4 = idx & 31;
    *(v4f*)(wsMP + (size_t)(g0 + s) * DD + d4 * 4) =
        *(const v4f*)(s_Proj + (s * 2 + 1) * PS + d4 * 4);
  }
}

// ---------------------------------------------------------------------------
// Kernel 2: per-b masked softmax over N + weighted sum of mp, relu.
// Block = 128 threads (wave h handles head h), one block per b.
__global__ __launch_bounds__(128)
void intrama_k2(const float* __restrict__ wsA, const float* __restrict__ wsMP,
                const int* __restrict__ masks, float* __restrict__ out) {
  __shared__ float s_am[HH][NN];
  const int b  = blockIdx.x;
  const int t  = threadIdx.x;   // t = h*32 + dd
  const int h  = t >> 5;
  const int dd = t & 31;

  float mx = -3.0e38f;
  for (int n = dd; n < NN; n += 32) {   // wsA transposed: contiguous in n
    float av = (masks[b * NN + n] == 1) ? wsA[(size_t)h * BN + b * NN + n] : 0.f;
    s_am[h][n] = av;
    mx = fmaxf(mx, av);
  }
#pragma unroll
  for (int off = 16; off; off >>= 1) mx = fmaxf(mx, __shfl_xor(mx, off, 32));
  __syncthreads();

  float se = 0.f;
  for (int n = dd; n < NN; n += 32) se += __expf(s_am[h][n] - mx);
#pragma unroll
  for (int off = 16; off; off >>= 1) se += __shfl_xor(se, off, 32);
  float inv = 1.f / se;

  float acc = 0.f;
  for (int n = 0; n < NN; ++n) {
    float w = __expf(s_am[h][n] - mx) * inv;
    acc += w * wsMP[((size_t)b * NN + n) * DD + t];  // coalesced over t
  }
  out[b * DD + t] = fmaxf(acc, 0.f);
}

// ---------------------------------------------------------------------------
extern "C" void kernel_launch(void* const* d_in, const int* in_sizes, int n_in,
                              void* d_out, int out_size, void* d_ws, size_t ws_size,
                              hipStream_t stream) {
  const float* instances = (const float*)d_in[0];
  const int*   masks     = (const int*)  d_in[1];
  const float* pos_emb   = (const float*)d_in[2];
  const float* in_w      = (const float*)d_in[3];
  const float* in_b      = (const float*)d_in[4];
  const float* out_w     = (const float*)d_in[5];
  const float* out_b     = (const float*)d_in[6];
  const float* attn1_w   = (const float*)d_in[7];
  const float* attn2v    = (const float*)d_in[8];
  float* out  = (float*)d_out;
  float* wsA  = (float*)d_ws;                 // HH*BN floats (gating logits, [h][bn])
  float* wsMP = wsA + (size_t)BN * HH;        // BN*128 floats (mp rows)

  intrama_k1<<<BN / GROUP, 256, 0, stream>>>(instances, pos_emb, in_w, in_b,
                                             out_w, out_b, attn1_w, attn2v,
                                             wsA, wsMP);
  intrama_k2<<<BB, DD, 0, stream>>>(wsA, wsMP, masks, out);
}
